// RITS_38637525794882
// MI455X (gfx1250) — matrix operand order repaired and checked by reading, not compiled, not run on hardware
//
#include <hip/hip_runtime.h>

typedef __attribute__((ext_vector_type(16))) _Float16 v16h;
typedef __attribute__((ext_vector_type(8)))  _Float16 v8h;
typedef __attribute__((ext_vector_type(8)))  float    v8f;
typedef __attribute__((ext_vector_type(4)))  float    v4f;
typedef __attribute__((ext_vector_type(4)))  unsigned int u32x4;
typedef __attribute__((ext_vector_type(8)))  int      i32x8;
typedef __attribute__((ext_vector_type(4)))  int      i32x4;

namespace {

constexpr int kN = 256, kT = 512, kD = 64, kH = 64, kMT = 16;
constexpr int kThreads = 128;

// ---- LDS byte offsets (dynamic shared memory carve-up) ----
// f16 pre-swizzled weight fragments (fragment = 32 lanes x 16 halves = 1KB)
constexpr int OFF_WH   = 0;                               // W_hist  64x64   8KB
constexpr int OFF_WF   = OFF_WH  + kH * kD * 2;           // W_feat  64x64   8KB
constexpr int OFF_WB   = OFF_WF  + kD * kD * 2;           // W_beta 128x64  16KB
constexpr int OFF_WGH  = OFF_WB  + 2 * kD * kD * 2;       // W_gh    64x64   8KB
constexpr int OFF_WL   = OFF_WGH + kD * kH * 2;           // W_lstm 128x256 64KB
constexpr int OFF_UL   = OFF_WL  + 2 * kD * 4 * kH * 2;   // U_lstm  64x256 32KB
// double-buffered TDM-staged x/m/d tiles: per buffer x(4KB) m(4KB) d(4KB)
constexpr int OFF_XB   = OFF_UL  + kH * 4 * kH * 2;       // 2 x 12KB
constexpr int kBufStride = kMT * kD * 4 * 3;              // 12288 B
// f32 activation staging
constexpr int OFF_XHAT = OFF_XB   + 2 * kBufStride;       // x_hat 16x64
constexpr int OFF_ZHAT = OFF_XHAT + kMT * kD * 4;         // z_hat
constexpr int OFF_BETA = OFF_ZHAT + kMT * kD * 4;         // beta
constexpr int OFF_SZ   = OFF_BETA + kMT * kD * 4;         // z gates 16x256 16KB
constexpr int OFF_SH   = OFF_SZ   + kMT * 4 * kH * 4;     // h state
constexpr int OFF_SC   = OFF_SH   + kMT * kH * 4;         // c state
// f16 A-operand staging (row-major)
constexpr int OFF_H16  = OFF_SC   + kMT * kH * 4;         // h (f16)
constexpr int OFF_HG16 = OFF_H16  + kMT * kH * 2;         // h * gamma_h (f16)
constexpr int OFF_D16  = OFF_HG16 + kMT * kH * 2;         // d (f16)
constexpr int OFF_XC16 = OFF_D16  + kMT * kD * 2;         // x_c (f16)
constexpr int OFF_SA   = OFF_XC16 + kMT * kD * 2;         // [gamma_x|m] / [c_c|m] 16x128
// f32 biases / diag
constexpr int OFF_BH   = OFF_SA   + kMT * 2 * kD * 2;
constexpr int OFF_BF   = OFF_BH   + kD * 4;
constexpr int OFF_BGX  = OFF_BF   + kD * 4;
constexpr int OFF_BGH  = OFF_BGX  + kD * 4;
constexpr int OFF_BBE  = OFF_BGH  + kH * 4;
constexpr int OFF_DGX  = OFF_BBE  + kD * 4;
constexpr int OFF_BL   = OFF_DGX  + kD * 4;
constexpr int SMEM_BYTES = OFF_BL + 4 * kH * 4;           // ~210.5 KB < 320 KB/WGP

__device__ __forceinline__ float sigm(float x) { return 1.0f / (1.0f + __expf(-x)); }

__device__ __forceinline__ v8f wmma_f16(v16h a, v16h b, v8f c) {
  // D = A(16x32 f16) * B(32x16 f16) + C(16x16 f32)
  return __builtin_amdgcn_wmma_f32_16x16x32_f16(false, a, false, b, (short)0, c,
                                                false, false);
}

// A operand: 16xK f16 row-major in LDS. CDNA5 16-bit A layout:
// lanes 0-15 (row=lane): halves 0-7 = K kb..kb+7, halves 8-15 = K kb+16..kb+23, kb=0
// lanes 16-31 (row=lane-16): same with kb=8.
__device__ __forceinline__ v16h load_a32(const _Float16* A, int ldk, int kchunk, int lane) {
  const int row = lane & 15;
  const int kb  = (lane >> 4) << 3;
  const _Float16* p = A + row * ldk + kchunk * 32 + kb;
  v8h lo = *(const v8h*)(p);
  v8h hi = *(const v8h*)(p + 16);
  return __builtin_shufflevector(lo, hi, 0, 1, 2, 3, 4, 5, 6, 7,
                                 8, 9, 10, 11, 12, 13, 14, 15);
}

// B operand fragment, pre-swizzled: contiguous 16 halves (32B) per lane.
__device__ __forceinline__ v16h load_b32(const _Float16* W, int nnt, int kt, int nt, int lane) {
  return *(const v16h*)(W + (((kt * nnt + nt) << 5) + lane) * 16);
}

// Swizzle a (K x Nw) row-major f32 weight into f16 B-fragment layout.
__device__ void swz(const float* __restrict__ W, int K, int Nw, _Float16* dst,
                    int zero_diag, int tid) {
  const int nnt = Nw >> 4;
  const int total = (K >> 5) * nnt * 32;   // frag-lanes
  for (int i = tid; i < total; i += kThreads) {
    const int l = i & 31;
    const int frag = i >> 5;
    const int kt = frag / nnt, nt = frag - kt * nnt;
    const int col = (nt << 4) + (l & 15);
    const int kr = (kt << 5) + ((l >> 4) << 3);
    _Float16* p = dst + i * 16;
#pragma unroll
    for (int j = 0; j < 16; ++j) {
      const int k = kr + j + ((j >> 3) << 3);
      float v = W[k * Nw + col];
      if (zero_diag && k == col) v = 0.0f;   // ZeroDiagonalConstraint (W_feat)
      p[j] = (_Float16)v;
    }
  }
}

// Tensor Data Mover: async 2D tile load Global -> LDS.
// Tile = 16 rows x 64 f32, row stride 3*T*D elements (batch-major input layout).
// D# built per CDNA5 ISA 8.3/8.4: group0 = {count|flags, lds_addr, global_addr, type},
// group1 = {mask/data_size, dims, tile dims, strides}. Tracked by TENSORcnt.
__device__ __forceinline__ void tdm_load_tile(const void* gsrc, unsigned lds_off) {
  const unsigned long long ga = (unsigned long long)(uintptr_t)gsrc;
  u32x4 g0;
  g0[0] = 1u;                                               // count=1 (valid user D#)
  g0[1] = lds_off;                                          // lds_addr (bytes)
  g0[2] = (unsigned)ga;                                     // global_addr[31:0]
  g0[3] = (unsigned)((ga >> 32) & 0x1ffffffull) | (2u << 30); // addr[56:32] | type=2
  i32x8 g1;
  g1[0] = (int)(2u << 16);          // workgroup_mask=0, data_size=2 (4 bytes)
  g1[1] = (int)(64u << 16);         // tensor_dim0[15:0]=64 (bits 63:48 of group)
  g1[2] = (int)(16u << 16);         // tensor_dim0 hi=0, tensor_dim1[15:0]=16
  g1[3] = (int)(64u << 16);         // tensor_dim1 hi=0, tile_dim0=64
  g1[4] = (int)(16u);               // tile_dim1=16, tile_dim2=0
  g1[5] = (int)(3 * kT * kD);       // tensor_dim0_stride[31:0] = 98304 elements
  g1[6] = 0;                        // stride hi=0, tensor_dim1_stride lo=0
  g1[7] = 0;
  i32x4 zz; zz[0] = 0; zz[1] = 0; zz[2] = 0; zz[3] = 0;     // groups 2/3 unused (2D)
  i32x8 z8;
#pragma unroll
  for (int j = 0; j < 8; ++j) z8[j] = 0;
  // 6-arg toolchain variant: (g0, g1, grp2, grp3, extra_grp, cpol)
  __builtin_amdgcn_tensor_load_to_lds(g0, g1, zz, zz, z8, 0);
}

__global__ void __launch_bounds__(kThreads)
rits_persistent(const float* __restrict__ inputs,
                const float* __restrict__ W_hist, const float* __restrict__ b_hist,
                const float* __restrict__ W_feat, const float* __restrict__ b_feat,
                const float* __restrict__ W_gx,   const float* __restrict__ b_gx,
                const float* __restrict__ W_gh,   const float* __restrict__ b_gh,
                const float* __restrict__ W_beta, const float* __restrict__ b_beta,
                const float* __restrict__ W_lstm, const float* __restrict__ U_lstm,
                const float* __restrict__ b_lstm,
                const float* __restrict__ W_out,  const float* __restrict__ b_out,
                float* __restrict__ imput, float* __restrict__ preds) {
  extern __shared__ char smem[];
  _Float16* Wh_l  = (_Float16*)(smem + OFF_WH);
  _Float16* Wf_l  = (_Float16*)(smem + OFF_WF);
  _Float16* Wb_l  = (_Float16*)(smem + OFF_WB);
  _Float16* Wgh_l = (_Float16*)(smem + OFF_WGH);
  _Float16* Wl_l  = (_Float16*)(smem + OFF_WL);
  _Float16* Ul_l  = (_Float16*)(smem + OFF_UL);
  float* sxhat = (float*)(smem + OFF_XHAT);
  float* szhat = (float*)(smem + OFF_ZHAT);
  float* sbeta = (float*)(smem + OFF_BETA);
  float* sz    = (float*)(smem + OFF_SZ);
  float* sh    = (float*)(smem + OFF_SH);
  float* sc    = (float*)(smem + OFF_SC);
  _Float16* sh16  = (_Float16*)(smem + OFF_H16);
  _Float16* shg16 = (_Float16*)(smem + OFF_HG16);
  _Float16* sd16  = (_Float16*)(smem + OFF_D16);
  _Float16* sxc16 = (_Float16*)(smem + OFF_XC16);
  _Float16* sA    = (_Float16*)(smem + OFF_SA);
  float* bh_l  = (float*)(smem + OFF_BH);
  float* bf_l  = (float*)(smem + OFF_BF);
  float* bgx_l = (float*)(smem + OFF_BGX);
  float* bgh_l = (float*)(smem + OFF_BGH);
  float* bbe_l = (float*)(smem + OFF_BBE);
  float* dgx_l = (float*)(smem + OFF_DGX);
  float* bl_l  = (float*)(smem + OFF_BL);

  const int tid  = threadIdx.x;
  const int lane = tid & 31;
  const int wv   = tid >> 5;
  const int n0   = blockIdx.x * kMT;   // this block's 16 batch rows

  // ---- one-time prep: swizzle weights f32->f16 fragments, stage biases ----
  swz(W_hist, kH,     kD,     Wh_l,  0, tid);
  swz(W_feat, kD,     kD,     Wf_l,  1, tid);     // zero diagonal
  swz(W_beta, 2 * kD, kD,     Wb_l,  0, tid);
  swz(W_gh,   kD,     kH,     Wgh_l, 0, tid);
  swz(W_lstm, 2 * kD, 4 * kH, Wl_l,  0, tid);
  swz(U_lstm, kH,     4 * kH, Ul_l,  0, tid);
  for (int i = tid; i < kD; i += kThreads) {
    bh_l[i] = b_hist[i]; bf_l[i] = b_feat[i]; bgx_l[i] = b_gx[i];
    bgh_l[i] = b_gh[i];  bbe_l[i] = b_beta[i];
    dgx_l[i] = W_gx[i * kD + i];                  // DiagonalConstraint -> diag only
  }
  for (int i = tid; i < 4 * kH; i += kThreads) bl_l[i] = b_lstm[i];
  for (int i = tid; i < kMT * kH; i += kThreads) { sh[i] = 0.f; sc[i] = 0.f; }

  // ---- TDM: stage step-0 x/m/d tiles (wave 0 issues, waits TENSORcnt) ----
  if (wv == 0) {
    const size_t tb = (size_t)(n0 * 3) * kT * kD;   // inputs[n0][0][0][0]
    tdm_load_tile(inputs + tb,                       OFF_XB);
    tdm_load_tile(inputs + tb + (size_t)kT * kD,     OFF_XB + 4096);
    tdm_load_tile(inputs + tb + 2 * (size_t)kT * kD, OFF_XB + 8192);
    __builtin_amdgcn_s_wait_tensorcnt(0);
  }
  __syncthreads();

  const int colw = (wv << 4) + (lane & 15);   // this wave's output column (D=64 GEMMs)
  const int rb   = (lane >> 4) << 3;          // accumulator row base (C/D layout)
  const int e8   = tid << 3;                  // contiguous 8-elem chunk per thread
  const int erow = e8 >> 6, ecol = e8 & 63;

  for (int t = 0; t < kT; ++t) {
    const int cur = t & 1;
    const float* sx  = (const float*)(smem + OFF_XB + cur * kBufStride);
    const float* smk = sx + kMT * kD;
    const float* sd  = smk + kMT * kD;

    // ---- prefetch next step's x/m/d via TDM into the other buffer ----
    if (wv == 0 && t + 1 < kT) {
      const size_t tb = ((size_t)(n0 * 3) * kT + (t + 1)) * kD;
      const unsigned nb = (unsigned)(OFF_XB + (1 - cur) * kBufStride);
      tdm_load_tile(inputs + tb,                       nb);
      tdm_load_tile(inputs + tb + (size_t)kT * kD,     nb + 4096);
      tdm_load_tile(inputs + tb + 2 * (size_t)kT * kD, nb + 8192);
    }

    // ---- Phase A: f16 conversions + gamma_x (vectorized, 8 elems/thread) ----
    {
      v4f d0 = *(const v4f*)(sd + e8),  d1 = *(const v4f*)(sd + e8 + 4);
      v4f m0 = *(const v4f*)(smk + e8), m1 = *(const v4f*)(smk + e8 + 4);
      v4f h0 = *(const v4f*)(sh + e8),  h1 = *(const v4f*)(sh + e8 + 4);
      v8h dh, mh, hh, gx;
#pragma unroll
      for (int j = 0; j < 4; ++j) {
        dh[j] = (_Float16)d0[j]; dh[j + 4] = (_Float16)d1[j];
        mh[j] = (_Float16)m0[j]; mh[j + 4] = (_Float16)m1[j];
        hh[j] = (_Float16)h0[j]; hh[j + 4] = (_Float16)h1[j];
        gx[j]     = (_Float16)__expf(-fmaxf(d0[j] * dgx_l[ecol + j]     + bgx_l[ecol + j],     0.f));
        gx[j + 4] = (_Float16)__expf(-fmaxf(d1[j] * dgx_l[ecol + j + 4] + bgx_l[ecol + j + 4], 0.f));
      }
      *(v8h*)(sd16 + e8) = dh;
      *(v8h*)(sh16 + e8) = hh;
      *(v8h*)(sA + (erow << 7) + 64 + ecol) = mh;   // concat high half = m
      *(v8h*)(sA + (erow << 7) + ecol)      = gx;   // concat low half  = gamma_x
    }
    __syncthreads();

    // ---- Phase B: x_hat = h@W_hist + b ; x_c -> f16 ----
    {
      v8f acc = {};
      const v16h a0 = load_a32(sh16, kH, 0, lane);
      const v16h a1 = load_a32(sh16, kH, 1, lane);
      acc = wmma_f16(a0, load_b32(Wh_l, 4, 0, wv, lane), acc);
      acc = wmma_f16(a1, load_b32(Wh_l, 4, 1, wv, lane), acc);
      const float bb = bh_l[colw];
#pragma unroll
      for (int r = 0; r < 8; ++r) {
        const int row = rb + r;
        const float xh = acc[r] + bb;
        const int e = (row << 6) + colw;
        const float mv = smk[e], xv = sx[e];
        sxhat[e] = xh;
        sxc16[e] = (_Float16)(mv * xv + (1.f - mv) * xh);
      }
    }
    __syncthreads();

    // ---- Phase C: z_hat ; beta = sig([gx,m]@W_beta) ; gamma_h ; h*gh -> f16 ----
    {
      v8f acc = {};
      acc = wmma_f16(load_a32(sxc16, kD, 0, lane), load_b32(Wf_l, 4, 0, wv, lane), acc);
      acc = wmma_f16(load_a32(sxc16, kD, 1, lane), load_b32(Wf_l, 4, 1, wv, lane), acc);
      const float bb = bf_l[colw];
#pragma unroll
      for (int r = 0; r < 8; ++r)
        szhat[((rb + r) << 6) + colw] = acc[r] + bb;
      v8f accb = {};
#pragma unroll
      for (int kt = 0; kt < 4; ++kt)
        accb = wmma_f16(load_a32(sA, 2 * kD, kt, lane), load_b32(Wb_l, 4, kt, wv, lane), accb);
      v8f accg = {};
      accg = wmma_f16(load_a32(sd16, kD, 0, lane), load_b32(Wgh_l, 4, 0, wv, lane), accg);
      accg = wmma_f16(load_a32(sd16, kD, 1, lane), load_b32(Wgh_l, 4, 1, wv, lane), accg);
      const float bbv = bbe_l[colw];
      const float bg  = bgh_l[colw];
#pragma unroll
      for (int r = 0; r < 8; ++r) {
        const int row = rb + r;
        const int e = (row << 6) + colw;
        sbeta[e] = sigm(accb[r] + bbv);
        const float gh = __expf(-fmaxf(accg[r] + bg, 0.f));
        shg16[e] = (_Float16)(sh[e] * gh);
      }
    }
    __syncthreads();

    // ---- Phase E: c_hat, c_c ; write xh/zh/ch outputs coalesced (b128) ----
    {
      v4f be0 = *(const v4f*)(sbeta + e8), be1 = *(const v4f*)(sbeta + e8 + 4);
      v4f zh0 = *(const v4f*)(szhat + e8), zh1 = *(const v4f*)(szhat + e8 + 4);
      v4f xh0 = *(const v4f*)(sxhat + e8), xh1 = *(const v4f*)(sxhat + e8 + 4);
      v4f mv0 = *(const v4f*)(smk + e8),   mv1 = *(const v4f*)(smk + e8 + 4);
      v4f xv0 = *(const v4f*)(sx + e8),    xv1 = *(const v4f*)(sx + e8 + 4);
      v4f ch0, ch1;
      v8h ccv;
#pragma unroll
      for (int j = 0; j < 4; ++j) {
        const float ca = be0[j] * zh0[j] + (1.f - be0[j]) * xh0[j];
        ccv[j] = (_Float16)(mv0[j] * xv0[j] + (1.f - mv0[j]) * ca);
        ch0[j] = ca;
        const float cb = be1[j] * zh1[j] + (1.f - be1[j]) * xh1[j];
        ccv[j + 4] = (_Float16)(mv1[j] * xv1[j] + (1.f - mv1[j]) * cb);
        ch1[j] = cb;
      }
      *(v8h*)(sA + (erow << 7) + ecol) = ccv;      // concat low half = c_c
      const size_t ob = (((size_t)(n0 + erow) * 3) * kT + t) * kD + ecol;
      *(v4f*)(imput + ob)     = xh0;                               // imputation 0
      *(v4f*)(imput + ob + 4) = xh1;
      *(v4f*)(imput + ob + (size_t)kT * kD)     = zh0;             // imputation 1
      *(v4f*)(imput + ob + (size_t)kT * kD + 4) = zh1;
      *(v4f*)(imput + ob + 2 * (size_t)kT * kD)     = ch0;         // imputation 2
      *(v4f*)(imput + ob + 2 * (size_t)kT * kD + 4) = ch1;
    }
    __syncthreads();

    // ---- Phase F: z = [c_c,m]@W_lstm + (h*gh)@U_lstm + b  (each wave = 1 gate) ----
    {
      const v16h ac0 = load_a32(sA, 2 * kD, 0, lane);
      const v16h ac1 = load_a32(sA, 2 * kD, 1, lane);
      const v16h ac2 = load_a32(sA, 2 * kD, 2, lane);
      const v16h ac3 = load_a32(sA, 2 * kD, 3, lane);
      const v16h ah0 = load_a32(shg16, kH, 0, lane);
      const v16h ah1 = load_a32(shg16, kH, 1, lane);
#pragma unroll
      for (int ii = 0; ii < 4; ++ii) {
        const int nt = (wv << 2) + ii;
        v8f acc = {};
        acc = wmma_f16(ac0, load_b32(Wl_l, 16, 0, nt, lane), acc);
        acc = wmma_f16(ac1, load_b32(Wl_l, 16, 1, nt, lane), acc);
        acc = wmma_f16(ac2, load_b32(Wl_l, 16, 2, nt, lane), acc);
        acc = wmma_f16(ac3, load_b32(Wl_l, 16, 3, nt, lane), acc);
        acc = wmma_f16(ah0, load_b32(Ul_l, 16, 0, nt, lane), acc);
        acc = wmma_f16(ah1, load_b32(Ul_l, 16, 1, nt, lane), acc);
        const int zc = (nt << 4) + (lane & 15);
        const float bz = bl_l[zc];
#pragma unroll
        for (int r = 0; r < 8; ++r)
          sz[((rb + r) << 8) + zc] = acc[r] + bz;
      }
    }
    __syncthreads();

    // ---- Phase G: LSTM state update (gates i,f,g,o), vectorized ----
    {
      const int zb = (erow << 8) + ecol;
      v4f gi0 = *(const v4f*)(sz + zb),       gi1 = *(const v4f*)(sz + zb + 4);
      v4f gf0 = *(const v4f*)(sz + zb + 64),  gf1 = *(const v4f*)(sz + zb + 68);
      v4f gg0 = *(const v4f*)(sz + zb + 128), gg1 = *(const v4f*)(sz + zb + 132);
      v4f go0 = *(const v4f*)(sz + zb + 192), go1 = *(const v4f*)(sz + zb + 196);
      v4f c0 = *(const v4f*)(sc + e8), c1 = *(const v4f*)(sc + e8 + 4);
      v4f cn0, cn1, hn0, hn1;
#pragma unroll
      for (int j = 0; j < 4; ++j) {
        cn0[j] = sigm(gf0[j]) * c0[j] + sigm(gi0[j]) * tanhf(gg0[j]);
        hn0[j] = sigm(go0[j]) * tanhf(cn0[j]);
        cn1[j] = sigm(gf1[j]) * c1[j] + sigm(gi1[j]) * tanhf(gg1[j]);
        hn1[j] = sigm(go1[j]) * tanhf(cn1[j]);
      }
      *(v4f*)(sc + e8)     = cn0; *(v4f*)(sc + e8 + 4) = cn1;
      *(v4f*)(sh + e8)     = hn0; *(v4f*)(sh + e8 + 4) = hn1;
    }
    // TDM prefetch for t+1 must land before next iteration reads it.
    if (wv == 0) __builtin_amdgcn_s_wait_tensorcnt(0);
    __syncthreads();
  }

  // ---- predictions = sigmoid(h_last @ W_out + b_out) ----
  if (tid < kMT) {
    float s = b_out[0];
    for (int k = 0; k < kH; ++k) s += sh[(tid << 6) + k] * W_out[k];
    preds[n0 + tid] = sigm(s);
  }
}

}  // namespace

extern "C" void kernel_launch(void* const* d_in, const int* in_sizes, int n_in,
                              void* d_out, int out_size, void* d_ws, size_t ws_size,
                              hipStream_t stream) {
  (void)in_sizes; (void)n_in; (void)out_size; (void)d_ws; (void)ws_size;
  const float* inputs = (const float*)d_in[0];
  const float* W_hist = (const float*)d_in[1];
  const float* b_hist = (const float*)d_in[2];
  const float* W_feat = (const float*)d_in[3];
  const float* b_feat = (const float*)d_in[4];
  const float* W_gx   = (const float*)d_in[5];
  const float* b_gx   = (const float*)d_in[6];
  const float* W_gh   = (const float*)d_in[7];
  const float* b_gh   = (const float*)d_in[8];
  const float* W_beta = (const float*)d_in[9];
  const float* b_beta = (const float*)d_in[10];
  const float* W_lstm = (const float*)d_in[11];
  const float* U_lstm = (const float*)d_in[12];
  const float* b_lstm = (const float*)d_in[13];
  const float* W_out  = (const float*)d_in[14];
  const float* b_out  = (const float*)d_in[15];

  float* imput = (float*)d_out;                          // (N,3,T,D) f32
  float* preds = imput + (size_t)kN * 3 * kT * kD;       // (N,1) f32, concatenated

  (void)hipFuncSetAttribute((const void*)rits_persistent,
                            hipFuncAttributeMaxDynamicSharedMemorySize, SMEM_BYTES);

  rits_persistent<<<dim3(kN / kMT), dim3(kThreads), SMEM_BYTES, stream>>>(
      inputs, W_hist, b_hist, W_feat, b_feat, W_gx, b_gx, W_gh, b_gh,
      W_beta, b_beta, W_lstm, U_lstm, b_lstm, W_out, b_out, imput, preds);
}
